// Mamba2Layer_87076166960158
// MI455X (gfx1250) — matrix operand order repaired and checked by reading, not compiled, not run on hardware
//
#include <hip/hip_runtime.h>
#include <hip/hip_bf16.h>

// ---------------------------------------------------------------------------
// Mamba2 layer for MI455X (gfx1250, wave32, WMMA 16x16x32 bf16)
//
// Sizes (from reference): B=4, L=4096, HID=1152, D_INNER=2304, D_STATE=64,
// HEADDIM=64, NHEADS=36, CHUNK=64, D_CONV=4, CONV_DIM=2432, D_IN_PROJ=4772.
// ---------------------------------------------------------------------------

#define BATCH      4
#define SEQLEN     4096
#define HID        1152
#define D_INNER    2304
#define D_STATE    64
#define HEADDIM    64
#define NHEADS     36
#define CHUNK      64
#define NCHUNK     (SEQLEN / CHUNK)          // 64
#define CONV_DIM   (D_INNER + 2 * D_STATE)   // 2432
#define D_IN_PROJ  (2 * D_INNER + 2 * D_STATE + NHEADS) // 4772
#define NROWS      (BATCH * SEQLEN)          // 16384

typedef __attribute__((ext_vector_type(16))) __bf16 v16bf;
typedef __attribute__((ext_vector_type(8)))  float  v8f;
typedef __attribute__((ext_vector_type(4)))  float  v4f;

__device__ __forceinline__ v8f wmma_bf16(v16bf a, v16bf b, v8f c) {
  // (neg_a, A, neg_b, B, c_mod, C, reuse_a, reuse_b)
  return __builtin_amdgcn_wmma_f32_16x16x32_bf16(false, a, false, b,
                                                 (short)0, c, false, false);
}

// Convert f32 p[0..7] and p[16..23] (16B-aligned) -> bf16 fragment (one b128 load x4).
__device__ __forceinline__ v16bf cvt16(const float* p) {
  v4f f0 = *(const v4f*)(p);
  v4f f1 = *(const v4f*)(p + 4);
  v4f f2 = *(const v4f*)(p + 16);
  v4f f3 = *(const v4f*)(p + 20);
  v16bf b;
#pragma unroll
  for (int i = 0; i < 4; ++i) {
    b[i]      = (__bf16)f0[i];
    b[4 + i]  = (__bf16)f1[i];
    b[8 + i]  = (__bf16)f2[i];
    b[12 + i] = (__bf16)f3[i];
  }
  return b;
}

// A fragment 16x32 (MxK) from row-major f32, base = &A[m0][k0], stride lda.
// Lane layout: row = lane%16, kb = (lane<16 ? 0 : 8);
// elems [kb..kb+7] then [16+kb..16+kb+7] (two 32B runs, both 16B-aligned).
__device__ __forceinline__ v16bf loadA_rm(const float* base, int lda, int lane) {
  int r  = lane & 15;
  int kb = (lane & 16) ? 8 : 0;
  const float* p = base + (size_t)r * lda + kb;
  v4f f0 = *(const v4f*)(p);
  v4f f1 = *(const v4f*)(p + 4);
  v4f f2 = *(const v4f*)(p + 16);
  v4f f3 = *(const v4f*)(p + 20);
  v16bf a;
#pragma unroll
  for (int i = 0; i < 4; ++i) {
    a[i]      = (__bf16)f0[i];
    a[4 + i]  = (__bf16)f1[i];
    a[8 + i]  = (__bf16)f2[i];
    a[12 + i] = (__bf16)f3[i];
  }
  return a;
}

// ---------------------------------------------------------------------------
// Generic GEMM: O(MxN) = A(MxK, row-major) @ W(NxK, row-major)^T, bf16 WMMA.
// Block = 256 threads (8 waves); each wave -> 16(M) x 64(N) strip,
// block -> 128(M) x 64(N). grid = (ceil(N/64), M/128). K % 32 == 0.
// N-edge handled branch-free by clamping B row indices (WMMA column j of D
// depends only on column j of B; clamped lanes' columns are never stored).
// ---------------------------------------------------------------------------
__global__ void gemm_wt_bf16(const float* __restrict__ A,
                             const float* __restrict__ W,
                             float* __restrict__ O,
                             int M, int N, int K) {
  const int lane = threadIdx.x & 31;
  const int w    = threadIdx.x >> 5;
  const int m0   = blockIdx.y * 128 + w * 16;
  const int n0   = blockIdx.x * 64;
  const int r16  = lane & 15;
  const int kbB  = (lane & 16) ? 8 : 0;

  // Per-lane, K-invariant clamped B row offsets.
  int nn0 = n0 + r16;        nn0 = (nn0 < N) ? nn0 : (N - 1);
  int nn1 = n0 + 16 + r16;   nn1 = (nn1 < N) ? nn1 : (N - 1);
  int nn2 = n0 + 32 + r16;   nn2 = (nn2 < N) ? nn2 : (N - 1);
  int nn3 = n0 + 48 + r16;   nn3 = (nn3 < N) ? nn3 : (N - 1);
  const size_t ro0 = (size_t)nn0 * K + kbB;
  const size_t ro1 = (size_t)nn1 * K + kbB;
  const size_t ro2 = (size_t)nn2 * K + kbB;
  const size_t ro3 = (size_t)nn3 * K + kbB;

  v8f acc0 = {}, acc1 = {}, acc2 = {}, acc3 = {};
  const float* Arow = A + (size_t)m0 * K;
  for (int k0 = 0; k0 < K; k0 += 32) {
    __builtin_prefetch(Arow + (size_t)r16 * K + k0 + 64, 0, 1);
    v16bf a  = loadA_rm(Arow + k0, K, lane);
    v16bf b0 = cvt16(W + ro0 + k0);
    v16bf b1 = cvt16(W + ro1 + k0);
    v16bf b2 = cvt16(W + ro2 + k0);
    v16bf b3 = cvt16(W + ro3 + k0);
    acc0 = wmma_bf16(a, b0, acc0);
    acc1 = wmma_bf16(a, b1, acc1);
    acc2 = wmma_bf16(a, b2, acc2);
    acc3 = wmma_bf16(a, b3, acc3);
  }

  const int hi = (lane >> 4) & 1;
  v8f accs[4] = {acc0, acc1, acc2, acc3};
#pragma unroll
  for (int j = 0; j < 4; ++j) {
    int n = n0 + 16 * j + r16;
    if (n < N) {
#pragma unroll
      for (int r = 0; r < 8; ++r) {
        int m = m0 + r + 8 * hi;
        O[(size_t)m * N + n] = accs[j][r];
      }
    }
  }
}

// ---------------------------------------------------------------------------
// Depthwise causal conv (D_CONV=4) + bias + SiLU over the xBC slice of zxbcdt.
// ---------------------------------------------------------------------------
__global__ void conv_silu_kernel(const float* __restrict__ zx,
                                 const float* __restrict__ cw,
                                 const float* __restrict__ cb,
                                 float* __restrict__ xbc) {
  size_t idx = (size_t)blockIdx.x * 256 + threadIdx.x;
  if (idx >= (size_t)NROWS * CONV_DIM) return;
  int cix = (int)(idx % CONV_DIM);
  int row = (int)(idx / CONV_DIM);
  int l   = row & (SEQLEN - 1);
  float acc = cb[cix];
#pragma unroll
  for (int j = 0; j < 4; ++j) {
    int ls = l - 3 + j;
    if (ls >= 0)
      acc += zx[(size_t)(row - 3 + j) * D_IN_PROJ + D_INNER + cix] * cw[cix * 4 + j];
  }
  xbc[idx] = acc / (1.0f + __expf(-acc));   // SiLU
}

// ---------------------------------------------------------------------------
// dt = softplus(raw + dt_bias); dA = -exp(A_log) * dt
// ---------------------------------------------------------------------------
__global__ void dt_kernel(const float* __restrict__ zx,
                          const float* __restrict__ dt_bias,
                          const float* __restrict__ A_log,
                          float* __restrict__ dtw,
                          float* __restrict__ dAw) {
  int idx = blockIdx.x * 256 + threadIdx.x;
  if (idx >= NROWS * NHEADS) return;
  int hh  = idx % NHEADS;
  int row = idx / NHEADS;
  float xv = zx[(size_t)row * D_IN_PROJ + (D_IN_PROJ - NHEADS) + hh] + dt_bias[hh];
  float sp = (xv > 20.0f) ? xv : log1pf(__expf(xv));
  dtw[idx] = sp;
  dAw[idx] = -__expf(A_log[hh]) * sp;
}

// ---------------------------------------------------------------------------
// Per-(b, chunk, head) SSD intra-chunk work, all matmuls via WMMA:
//   G = Cc @ Bc^T ; M = mask(G)*exp(acum[l]-acum[s])
//   Y_diag = M @ (x*dt) ; states = (x*dt*decay)^T @ Bc
// 128 threads = 4 waves; wave w owns rows [16w, 16w+16) of the 64x64 outputs.
// ---------------------------------------------------------------------------
__global__ void ssd_chunk_kernel(const float* __restrict__ xbc,
                                 const float* __restrict__ dtw,
                                 const float* __restrict__ dAw,
                                 const float* __restrict__ Dparam,
                                 float* __restrict__ yout,
                                 float* __restrict__ states,
                                 float* __restrict__ casum,
                                 float* __restrict__ cacum) {
  __shared__ float xraw[CHUNK * HEADDIM];  // x chunk (unscaled)
  __shared__ float Bs[CHUNK * D_STATE];    // B chunk
  __shared__ float CM[CHUNK * CHUNK];      // Cc, then masked M
  __shared__ float adt[CHUNK], dts[CHUNK], acum[CHUNK], decs[CHUNK];

  const int h = blockIdx.x, c = blockIdx.y, b = blockIdx.z;
  const int tid = threadIdx.x;
  const size_t gbase = (size_t)b * SEQLEN + (size_t)c * CHUNK;
  const size_t cb = ((size_t)b * NCHUNK + c) * NHEADS + h;

  for (int i = tid; i < CHUNK * 64; i += 128) {
    int l = i >> 6, p = i & 63;
    const float* rowp = xbc + (gbase + l) * CONV_DIM;
    xraw[i] = rowp[h * HEADDIM + p];
    Bs[i]   = rowp[D_INNER + p];
    CM[i]   = rowp[D_INNER + D_STATE + p];
  }
  if (tid < CHUNK) {
    adt[tid] = dAw[(gbase + tid) * NHEADS + h];
    dts[tid] = dtw[(gbase + tid) * NHEADS + h];
  }
  __syncthreads();
  if (tid < CHUNK) {
    float s = 0.0f;
    for (int j = 0; j <= tid; ++j) s += adt[j];
    acum[tid] = s;
  }
  __syncthreads();
  if (tid < CHUNK) {
    decs[tid] = __expf(acum[CHUNK - 1] - acum[tid]);
    cacum[cb * CHUNK + tid] = acum[tid];
    if (tid == 0) casum[cb] = acum[CHUNK - 1];
  }
  __syncthreads();

  const int lane = tid & 31, w = tid >> 5;
  const int m0 = w * 16;
  const int r16 = lane & 15, hi = (lane >> 4) & 1;
  const int kbA = hi ? 8 : 0, kbB = hi ? 8 : 0;

  // ---- G = Cc @ Bc^T (K = D_STATE = 64) ----
  v8f g0 = {}, g1 = {}, g2 = {}, g3 = {};
#pragma unroll
  for (int kk = 0; kk < 64; kk += 32) {
    v16bf aC = loadA_rm(&CM[(m0 + r16) * 64 + kk], 0, lane & 16);  // lda unused: row folded in
    // NOTE: loadA_rm adds (lane&15)*lda; we already indexed the row, so pass
    // lane&16 only to select kb. Equivalent explicit form below for clarity:
    {
      const float* p = &CM[(m0 + r16) * 64 + kk + kbA];
      v4f f0 = *(const v4f*)(p);
      v4f f1 = *(const v4f*)(p + 4);
      v4f f2 = *(const v4f*)(p + 16);
      v4f f3 = *(const v4f*)(p + 20);
#pragma unroll
      for (int i = 0; i < 4; ++i) {
        aC[i] = (__bf16)f0[i]; aC[4+i] = (__bf16)f1[i];
        aC[8+i] = (__bf16)f2[i]; aC[12+i] = (__bf16)f3[i];
      }
    }
    v16bf bb0 = cvt16(&Bs[(0  + r16) * 64 + kk + kbB]);
    v16bf bb1 = cvt16(&Bs[(16 + r16) * 64 + kk + kbB]);
    v16bf bb2 = cvt16(&Bs[(32 + r16) * 64 + kk + kbB]);
    v16bf bb3 = cvt16(&Bs[(48 + r16) * 64 + kk + kbB]);
    g0 = wmma_bf16(aC, bb0, g0);
    g1 = wmma_bf16(aC, bb1, g1);
    g2 = wmma_bf16(aC, bb2, g2);
    g3 = wmma_bf16(aC, bb3, g3);
  }
  // Apply causal decay mask; overwrite CM (wave-local rows) with M.
  {
    v8f gg[4] = {g0, g1, g2, g3};
#pragma unroll
    for (int j = 0; j < 4; ++j) {
#pragma unroll
      for (int r = 0; r < 8; ++r) {
        int l = m0 + r + 8 * hi;
        int s = 16 * j + r16;
        CM[l * 64 + s] = (l >= s) ? gg[j][r] * __expf(acum[l] - acum[s]) : 0.0f;
      }
    }
  }
  __syncthreads();

  // ---- Y_diag = M @ (x * dt) ; then + D*x, store to y ----
  const float Dp = Dparam[h];
  v8f ya[4] = {};
#pragma unroll
  for (int kk = 0; kk < 64; kk += 32) {
    v16bf aM;
    {
      const float* p = &CM[(m0 + r16) * 64 + kk + kbA];
      v4f f0 = *(const v4f*)(p);
      v4f f1 = *(const v4f*)(p + 4);
      v4f f2 = *(const v4f*)(p + 16);
      v4f f3 = *(const v4f*)(p + 20);
#pragma unroll
      for (int i = 0; i < 4; ++i) {
        aM[i] = (__bf16)f0[i]; aM[4+i] = (__bf16)f1[i];
        aM[8+i] = (__bf16)f2[i]; aM[12+i] = (__bf16)f3[i];
      }
    }
#pragma unroll
    for (int j = 0; j < 4; ++j) {
      v16bf bX;
#pragma unroll
      for (int i = 0; i < 16; ++i) {
        int srow = kk + kbB + (i & 7) + ((i >> 3) << 4);
        bX[i] = (__bf16)(xraw[srow * 64 + 16 * j + r16] * dts[srow]);
      }
      ya[j] = wmma_bf16(aM, bX, ya[j]);
    }
  }
#pragma unroll
  for (int j = 0; j < 4; ++j) {
#pragma unroll
    for (int r = 0; r < 8; ++r) {
      int l = m0 + r + 8 * hi;
      int p = 16 * j + r16;
      yout[(gbase + l) * D_INNER + h * HEADDIM + p] = ya[j][r] + Dp * xraw[l * 64 + p];
    }
  }

  // ---- states[p][n] = sum_l x[l][p]*dt[l]*decay[l] * B[l][n] ----
  v8f sa[4] = {};
#pragma unroll
  for (int kk = 0; kk < 64; kk += 32) {
    v16bf aX;
#pragma unroll
    for (int i = 0; i < 8; ++i) {
      int l1 = kk + kbA + i;
      int l2 = kk + 16 + kbA + i;
      aX[i]     = (__bf16)(xraw[l1 * 64 + m0 + r16] * dts[l1] * decs[l1]);
      aX[i + 8] = (__bf16)(xraw[l2 * 64 + m0 + r16] * dts[l2] * decs[l2]);
    }
#pragma unroll
    for (int j = 0; j < 4; ++j) {
      v16bf bB;
#pragma unroll
      for (int i = 0; i < 16; ++i) {
        int l = kk + kbB + (i & 7) + ((i >> 3) << 4);
        bB[i] = (__bf16)Bs[l * 64 + 16 * j + r16];
      }
      sa[j] = wmma_bf16(aX, bB, sa[j]);
    }
  }
  const size_t sb = cb * (HEADDIM * D_STATE);
#pragma unroll
  for (int j = 0; j < 4; ++j) {
#pragma unroll
    for (int r = 0; r < 8; ++r)
      states[sb + (size_t)(m0 + r + 8 * hi) * 64 + 16 * j + r16] = sa[j][r];
  }
}

// ---------------------------------------------------------------------------
// Sequential inter-chunk recurrence per (b, h):
//   out[c] = carry ; carry = carry * exp(asum_c) + states_c
// 256 threads x 16 state elems each = 4096-element state in registers.
// ---------------------------------------------------------------------------
__global__ void state_scan_kernel(const float* __restrict__ states,
                                  const float* __restrict__ casum,
                                  float* __restrict__ stout) {
  const int hh = blockIdx.x, b = blockIdx.y;
  const int tid = threadIdx.x;
  float carry[16];
#pragma unroll
  for (int i = 0; i < 16; ++i) carry[i] = 0.0f;
  for (int c = 0; c < NCHUNK; ++c) {
    size_t cb = ((size_t)b * NCHUNK + c) * NHEADS + hh;
    size_t sb = cb * (HEADDIM * D_STATE) + (size_t)tid * 16;
    float ae = __expf(casum[cb]);
#pragma unroll
    for (int i = 0; i < 16; ++i) {
      float st = states[sb + i];
      stout[sb + i] = carry[i];
      carry[i] = carry[i] * ae + st;
    }
  }
}

// ---------------------------------------------------------------------------
// Y_off += (Cc * exp(acum[l])) @ states_in^T, per (b, chunk, head).
// ---------------------------------------------------------------------------
__global__ void yoff_kernel(const float* __restrict__ xbc,
                            const float* __restrict__ stout,
                            const float* __restrict__ cacum,
                            float* __restrict__ yout) {
  __shared__ float Cs[CHUNK * D_STATE];
  __shared__ float Sp[HEADDIM * D_STATE];
  const int h = blockIdx.x, c = blockIdx.y, b = blockIdx.z;
  const int tid = threadIdx.x;
  const size_t gbase = (size_t)b * SEQLEN + (size_t)c * CHUNK;
  const size_t cb = ((size_t)b * NCHUNK + c) * NHEADS + h;

  for (int i = tid; i < 4096; i += 128) {
    int l = i >> 6, n = i & 63;
    Cs[i] = xbc[(gbase + l) * CONV_DIM + D_INNER + D_STATE + n] *
            __expf(cacum[cb * CHUNK + l]);
    Sp[i] = stout[cb * (HEADDIM * D_STATE) + i];
  }
  __syncthreads();

  const int lane = tid & 31, w = tid >> 5;
  const int m0 = w * 16;
  const int r16 = lane & 15, hi = (lane >> 4) & 1;
  const int kbA = hi ? 8 : 0, kbB = hi ? 8 : 0;

  v8f acc0 = {}, acc1 = {}, acc2 = {}, acc3 = {};
#pragma unroll
  for (int kk = 0; kk < 64; kk += 32) {
    v16bf aC;
    {
      const float* p = &Cs[(m0 + r16) * 64 + kk + kbA];
      v4f f0 = *(const v4f*)(p);
      v4f f1 = *(const v4f*)(p + 4);
      v4f f2 = *(const v4f*)(p + 16);
      v4f f3 = *(const v4f*)(p + 20);
#pragma unroll
      for (int i = 0; i < 4; ++i) {
        aC[i] = (__bf16)f0[i]; aC[4+i] = (__bf16)f1[i];
        aC[8+i] = (__bf16)f2[i]; aC[12+i] = (__bf16)f3[i];
      }
    }
    v16bf bs0 = cvt16(&Sp[(0  + r16) * 64 + kk + kbB]);
    v16bf bs1 = cvt16(&Sp[(16 + r16) * 64 + kk + kbB]);
    v16bf bs2 = cvt16(&Sp[(32 + r16) * 64 + kk + kbB]);
    v16bf bs3 = cvt16(&Sp[(48 + r16) * 64 + kk + kbB]);
    acc0 = wmma_bf16(aC, bs0, acc0);
    acc1 = wmma_bf16(aC, bs1, acc1);
    acc2 = wmma_bf16(aC, bs2, acc2);
    acc3 = wmma_bf16(aC, bs3, acc3);
  }
  v8f accs[4] = {acc0, acc1, acc2, acc3};
#pragma unroll
  for (int j = 0; j < 4; ++j) {
#pragma unroll
    for (int r = 0; r < 8; ++r) {
      int l = m0 + r + 8 * hi;
      int p = 16 * j + r16;
      yout[(gbase + l) * D_INNER + h * HEADDIM + p] += accs[j][r];
    }
  }
}

// ---------------------------------------------------------------------------
// yg = y * silu(z); RMS-norm over D_INNER; scale by norm_w. One block per row.
// ---------------------------------------------------------------------------
__global__ void gate_norm_kernel(const float* __restrict__ y,
                                 const float* __restrict__ zx,
                                 const float* __restrict__ norm_w,
                                 float* __restrict__ out) {
  __shared__ float red[256];
  const size_t row = blockIdx.x;
  const int tid = threadIdx.x;
  float g[9];
  float ss = 0.0f;
#pragma unroll
  for (int it = 0; it < 9; ++it) {               // 2304 = 9 * 256
    int i = it * 256 + tid;
    float zv = zx[row * D_IN_PROJ + i];
    float gv = y[row * D_INNER + i] * zv / (1.0f + __expf(-zv));
    g[it] = gv;
    ss += gv * gv;
  }
  red[tid] = ss;
  __syncthreads();
  for (int s = 128; s > 0; s >>= 1) {
    if (tid < s) red[tid] += red[tid + s];
    __syncthreads();
  }
  float inv = rsqrtf(red[0] / (float)D_INNER + 1e-5f);
#pragma unroll
  for (int it = 0; it < 9; ++it) {
    int i = it * 256 + tid;
    out[row * D_INNER + i] = g[it] * inv * norm_w[i];
  }
}

// ---------------------------------------------------------------------------
extern "C" void kernel_launch(void* const* d_in, const int* in_sizes, int n_in,
                              void* d_out, int out_size, void* d_ws, size_t ws_size,
                              hipStream_t stream) {
  (void)in_sizes; (void)n_in; (void)out_size; (void)ws_size;
  const float* u          = (const float*)d_in[0];
  const float* in_proj_w  = (const float*)d_in[1];
  const float* conv_w     = (const float*)d_in[2];
  const float* conv_b     = (const float*)d_in[3];
  const float* dt_bias    = (const float*)d_in[4];
  const float* A_log      = (const float*)d_in[5];
  const float* D_param    = (const float*)d_in[6];
  const float* norm_w     = (const float*)d_in[7];
  const float* out_proj_w = (const float*)d_in[8];
  float* out = (float*)d_out;

  float* ws = (float*)d_ws;
  size_t off = 0;
  float* zx    = ws + off; off += (size_t)NROWS * D_IN_PROJ;             // 78.2M
  float* xbc   = ws + off; off += (size_t)NROWS * CONV_DIM;              // 39.8M
  float* dtw   = ws + off; off += (size_t)NROWS * NHEADS;
  float* dAw   = ws + off; off += (size_t)NROWS * NHEADS;
  float* yw    = ws + off; off += (size_t)NROWS * D_INNER;               // 37.7M
  float* st    = ws + off; off += (size_t)BATCH * NCHUNK * NHEADS * 4096;// 37.7M
  float* stout = ws + off; off += (size_t)BATCH * NCHUNK * NHEADS * 4096;// 37.7M
  float* casum = ws + off; off += (size_t)BATCH * NCHUNK * NHEADS;
  float* cacum = ws + off; off += (size_t)BATCH * NCHUNK * NHEADS * CHUNK;
  float* normed = st;  // states buffer is dead after the scan -> reuse

  // 1) in_proj GEMM: zx(16384 x 4772) = u(16384 x 1152) @ in_proj_w^T
  gemm_wt_bf16<<<dim3((D_IN_PROJ + 63) / 64, NROWS / 128), 256, 0, stream>>>(
      u, in_proj_w, zx, NROWS, D_IN_PROJ, HID);

  // 2) depthwise conv + SiLU on xBC slice
  conv_silu_kernel<<<(NROWS * CONV_DIM + 255) / 256, 256, 0, stream>>>(
      zx, conv_w, conv_b, xbc);

  // 3) dt / dA
  dt_kernel<<<(NROWS * NHEADS + 255) / 256, 256, 0, stream>>>(
      zx, dt_bias, A_log, dtw, dAw);

  // 4) intra-chunk SSD (WMMA): Y_diag + D*x, chunk states, decays
  ssd_chunk_kernel<<<dim3(NHEADS, NCHUNK, BATCH), 128, 0, stream>>>(
      xbc, dtw, dAw, D_param, yw, st, casum, cacum);

  // 5) inter-chunk state recurrence
  state_scan_kernel<<<dim3(NHEADS, BATCH), 256, 0, stream>>>(st, casum, stout);

  // 6) Y_off (WMMA) accumulated into y
  yoff_kernel<<<dim3(NHEADS, NCHUNK, BATCH), 128, 0, stream>>>(
      xbc, stout, cacum, yw);

  // 7) gate + RMS norm
  gate_norm_kernel<<<NROWS, 256, 0, stream>>>(yw, zx, norm_w, normed);

  // 8) out_proj GEMM: out(16384 x 1152) = normed(16384 x 2304) @ out_proj_w^T
  gemm_wt_bf16<<<dim3(HID / 64, NROWS / 128), 256, 0, stream>>>(
      normed, out_proj_w, out, NROWS, HID, D_INNER);
}
